// DiTBLockContinuous_61615600828423
// MI455X (gfx1250) — matrix operand order repaired and checked by reading, not compiled, not run on hardware
//
#include <hip/hip_runtime.h>
#include <hip/hip_bf16.h>
#include <cstdint>

// ---------------- types ----------------
typedef __bf16 bf16;
typedef __attribute__((ext_vector_type(16))) __bf16 v16bf;
typedef __attribute__((ext_vector_type(8)))  __bf16 v8bf;
typedef __attribute__((ext_vector_type(8)))  float  v8f;

#define B_     8
#define S_     1024
#define E_     1024
#define H_     16
#define D_     128
#define INNER_ 2048
#define FF_    4096
#define NTOK   8192

// CDNA5 async global->LDS path (ASYNCcnt-tracked, bypasses VGPRs)
#if defined(__has_builtin)
#if __has_builtin(__builtin_amdgcn_global_load_async_to_lds_b128) && \
    __has_builtin(__builtin_amdgcn_s_wait_asynccnt)
#define HAVE_ASYNC_LDS 1
#else
#define HAVE_ASYNC_LDS 0
#endif
#else
#define HAVE_ASYNC_LDS 0
#endif

#if HAVE_ASYNC_LDS
// builtin signature (from hipcc diagnostic): param0 = AS(1) v4i*, so mirror that;
// LDS destination uses AS(3).
typedef __attribute__((__vector_size__(4 * sizeof(int)))) int v4i_t;
typedef __attribute__((address_space(1))) v4i_t gv4i;
typedef __attribute__((address_space(3))) v4i_t lv4i;
#endif

// copy 16 bf16 (32B) from global to LDS, per-lane
__device__ __forceinline__ void stage16(const bf16* __restrict__ g, bf16* __restrict__ l) {
#if HAVE_ASYNC_LDS
  __builtin_amdgcn_global_load_async_to_lds_b128((gv4i*)(g),     (lv4i*)(l),     0, 0);
  __builtin_amdgcn_global_load_async_to_lds_b128((gv4i*)(g + 8), (lv4i*)(l + 8), 0, 0);
#else
  *(v8bf*)l       = *(const v8bf*)g;
  *(v8bf*)(l + 8) = *(const v8bf*)(g + 8);
#endif
}
__device__ __forceinline__ void async_drain() {
#if HAVE_ASYNC_LDS
  __builtin_amdgcn_s_wait_asynccnt(0);
#endif
}

__device__ __forceinline__ v8f wmma_bf16(v16bf a, v16bf b, v8f c) {
  // D = A(16x32 bf16) * B(32x16 bf16) + C(16x16 f32)
  return __builtin_amdgcn_wmma_f32_16x16x32_bf16(false, a, false, b, (short)0, c,
                                                 false, false);
}

__device__ __forceinline__ v16bf cat16(v8bf lo, v8bf hi) {
  v16bf r;
#pragma unroll
  for (int i = 0; i < 8; ++i) { r[i] = lo[i]; r[i + 8] = hi[i]; }
  return r;
}
__device__ __forceinline__ v8f vzero8() {
  v8f z;
#pragma unroll
  for (int i = 0; i < 8; ++i) z[i] = 0.f;
  return z;
}

// ---------------- fp32 -> bf16 weight conversion ----------------
__global__ __launch_bounds__(256) void f32_to_bf16_kernel(const float* __restrict__ src,
                                                          bf16* __restrict__ dst, int n) {
  int i = blockIdx.x * 256 + threadIdx.x;
  int stride = gridDim.x * 256;
  for (; i < n; i += stride) dst[i] = (bf16)src[i];
}

// ---------------- cond @ w_cond_{attn,glu} : (8,2048)x(2048,2048) ----------------
__global__ __launch_bounds__(256) void cond_gemm_kernel(const float* __restrict__ cond,
                                                        const float* __restrict__ wA,
                                                        const float* __restrict__ wG,
                                                        float* __restrict__ yA,
                                                        float* __restrict__ yG) {
  int gid   = blockIdx.x * 256 + threadIdx.x;   // 0..32767
  int which = gid >> 14;
  int idx   = gid & 16383;
  int row   = idx >> 11, col = idx & 2047;
  const float* w = which ? wG : wA;
  const float* c = cond + (size_t)row * 2 * E_;
  float acc = 0.f;
  for (int k = 0; k < 2 * E_; ++k) acc += c[k] * w[(size_t)k * 2 * E_ + col];
  (which ? yG : yA)[idx] = acc;
}

// ---------------- h = rms_norm(x)*(1+scale)+shift  -> bf16 ----------------
__global__ __launch_bounds__(256) void rms_mod_kernel(const float* __restrict__ x,
                                                      const float* __restrict__ y,
                                                      bf16* __restrict__ h) {
  int t = blockIdx.x;
  int b = t / S_;
  const float* xr = x + (size_t)t * E_;
  __shared__ float red[256];
  float ss = 0.f;
  for (int e = threadIdx.x; e < E_; e += 256) { float v = xr[e]; ss += v * v; }
  red[threadIdx.x] = ss;
  __syncthreads();
  for (int s = 128; s > 0; s >>= 1) {
    if (threadIdx.x < s) red[threadIdx.x] += red[threadIdx.x + s];
    __syncthreads();
  }
  float rn = rsqrtf(red[0] * (1.0f / E_) + 1e-6f);
  const float* yr = y + (size_t)b * 2 * E_;
  bf16* hr = h + (size_t)t * E_;
  for (int e = threadIdx.x; e < E_; e += 256)
    hr[e] = (bf16)(xr[e] * rn * (1.f + yr[e]) + yr[E_ + e]);
}

// ---------------- bf16 WMMA GEMM: C(MxN) = A(MxK) * B(KxN) [+res] ----------------
// BM=128 BN=128 BK=32, 256 threads = 8 waves, wave tile 64x32 = 4x2 WMMA accums.
// Double-buffered LDS; A tile staged via async global->LDS (ASYNCcnt),
// B tile transposed in-flight so B fragments are contiguous ds_load_b128s.
// MODE 0: store bf16. MODE 1: store f32 = acc + res.
template <int MODE>
__global__ __launch_bounds__(256) void gemm_bf16_kernel(const bf16* __restrict__ A,
                                                        const bf16* __restrict__ Bm,
                                                        void* __restrict__ outp,
                                                        const float* __restrict__ res,
                                                        int M, int N, int K) {
  const int tilesN = N >> 7;
  int tn = blockIdx.x % tilesN, tm = blockIdx.x / tilesN;
  int row0 = tm << 7, col0 = tn << 7;

  __shared__ __align__(16) bf16 sA[2][128 * 40];  // [m][k], pitch 40
  __shared__ __align__(16) bf16 sB[2][128 * 40];  // transposed [n][k], pitch 40

  int tid = threadIdx.x;
  int lane = tid & 31, wid = tid >> 5;
  int wm = (wid >> 2) * 64, wn = (wid & 3) * 32;
  int ln = lane & 15, half = lane >> 4;

  v8f acc[4][2];
#pragma unroll
  for (int i = 0; i < 4; ++i)
#pragma unroll
    for (int j = 0; j < 2; ++j) acc[i][j] = vzero8();

  auto stageA = [&](int k0, int buf) {
    int r = tid >> 1, seg = (tid & 1) * 16;
    stage16(A + (size_t)(row0 + r) * K + k0 + seg, &sA[buf][r * 40 + seg]);
  };
  auto stageB = [&](int k0, int buf) {
    int k = tid & 31, nseg = (tid >> 5) * 16;
    const bf16* src = Bm + (size_t)(k0 + k) * N + col0 + nseg;
    v8bf v0 = *(const v8bf*)(src);
    v8bf v1 = *(const v8bf*)(src + 8);
    bf16* dst = sB[buf];
#pragma unroll
    for (int j = 0; j < 8; ++j) dst[(nseg + j) * 40 + k]     = v0[j];
#pragma unroll
    for (int j = 0; j < 8; ++j) dst[(nseg + 8 + j) * 40 + k] = v1[j];
    if (k0 + 32 < K) __builtin_prefetch(src + (size_t)32 * N, 0, 1);
  };

  const int nk = K >> 5;
  stageA(0, 0);
  stageB(0, 0);

  for (int kt = 0; kt < nk; ++kt) {
    int buf = kt & 1;
    async_drain();
    __syncthreads();
    if (kt + 1 < nk) {  // prefetch next tile into the back buffer (overlaps WMMA)
      stageA((kt + 1) << 5, buf ^ 1);
      stageB((kt + 1) << 5, buf ^ 1);
    }

    v16bf af[4];
#pragma unroll
    for (int i = 0; i < 4; ++i) {
      int r = wm + i * 16 + ln;
      int kb = half * 8;
      af[i] = cat16(*(const v8bf*)(&sA[buf][r * 40 + kb]),
                    *(const v8bf*)(&sA[buf][r * 40 + 16 + kb]));
    }
#pragma unroll
    for (int j = 0; j < 2; ++j) {
      int n = wn + j * 16 + ln;
      int kb = half * 8;
      v16bf bq = cat16(*(const v8bf*)(&sB[buf][n * 40 + kb]),
                       *(const v8bf*)(&sB[buf][n * 40 + kb + 16]));
#pragma unroll
      for (int i = 0; i < 4; ++i) acc[i][j] = wmma_bf16(af[i], bq, acc[i][j]);
    }
  }

#pragma unroll
  for (int i = 0; i < 4; ++i)
#pragma unroll
    for (int j = 0; j < 2; ++j)
#pragma unroll
      for (int r = 0; r < 8; ++r) {
        int row = row0 + wm + i * 16 + r + 8 * half;
        int col = col0 + wn + j * 16 + ln;
        float v = acc[i][j][r];
        if (MODE == 0)
          ((bf16*)outp)[(size_t)row * N + col] = (bf16)v;
        else
          ((float*)outp)[(size_t)row * N + col] = v + res[(size_t)row * N + col];
      }
}

// ---------------- RoPE + relayout q/k/v -> [B,H,S,D] bf16 ----------------
__global__ __launch_bounds__(256) void rope_kernel(const bf16* __restrict__ qkv,
                                                   const float* __restrict__ pos,
                                                   bf16* __restrict__ qa,
                                                   bf16* __restrict__ ka,
                                                   bf16* __restrict__ va) {
  int t = blockIdx.x;           // token
  int b = t / S_, s = t % S_;
  const bf16* base = qkv + (size_t)t * (3 * INNER_);
  for (int i = threadIdx.x; i < H_ * D_; i += 256) {
    int hh = i >> 7, d = i & 127;
    int c0 = hh * D_ + d;
    float qv = (float)base[c0];
    float kv = (float)base[INNER_ + c0];
    float vv = (float)base[2 * INNER_ + c0];
    float sn = pos[(size_t)s * 256 + 2 * d];
    float cs = pos[(size_t)s * 256 + 2 * d + 1];
    int dr = (d < 64) ? d + 64 : d - 64;
    float sgn = (d < 64) ? -1.f : 1.f;
    float qr = (float)base[hh * D_ + dr];
    float kr = (float)base[INNER_ + hh * D_ + dr];
    size_t o = (((size_t)(b * H_ + hh)) * S_ + s) * D_ + d;
    qa[o] = (bf16)(qv * cs + sgn * qr * sn);
    ka[o] = (bf16)(kv * cs + sgn * kr * sn);
    va[o] = (bf16)vv;
  }
}

// ---------------- flash attention: 1 block per (b,h,q-block of 128) ----------------
__global__ __launch_bounds__(256) void attn_kernel(const bf16* __restrict__ Q,
                                                   const bf16* __restrict__ Kx,
                                                   const bf16* __restrict__ V,
                                                   bf16* __restrict__ O) {
  int qb = blockIdx.x & 7;
  int bh = blockIdx.x >> 3;
  int hh = bh & (H_ - 1), b = bh >> 4;
  const size_t base = (size_t)bh * S_ * D_;
  const bf16* q = Q + base + (size_t)qb * 128 * D_;
  const bf16* k = Kx + base;
  const bf16* v = V + base;

  __shared__ __align__(16) bf16 sK[64 * 136];   // [key][d]
  __shared__ __align__(16) bf16 sV[128 * 72];   // [d][key]
  __shared__ __align__(16) bf16 sP[8][16 * 72]; // per-wave P, [row][key]
  __shared__ float sM[8][16], sL[8][16], sMn[8][16], sAl[8][16];

  int tid = threadIdx.x, lane = tid & 31, wid = tid >> 5;
  int ln = lane & 15, half = lane >> 4;

  if (tid < 128) { sM[tid >> 4][tid & 15] = -3e30f; sL[tid >> 4][tid & 15] = 0.f; }

  // Q fragments live in registers for the whole kernel (4 x 32-wide K chunks of D=128)
  v16bf qf[4];
  {
    int r = wid * 16 + ln;
    int kb = half * 8;
#pragma unroll
    for (int c = 0; c < 4; ++c)
      qf[c] = cat16(*(const v8bf*)(q + (size_t)r * D_ + c * 32 + kb),
                    *(const v8bf*)(q + (size_t)r * D_ + c * 32 + 16 + kb));
  }
  v8f oacc[8];
#pragma unroll
  for (int dj = 0; dj < 8; ++dj) oacc[dj] = vzero8();
  const float scale = 0.08838834764831845f;  // 1/sqrt(128)

  for (int kb0 = 0; kb0 < S_; kb0 += 64) {
    __syncthreads();
    {  // stage K block (64x128) row-major — async global->LDS copy
      int r = tid >> 2, seg = (tid & 3) * 32;
      const bf16* src = k + (size_t)(kb0 + r) * D_ + seg;
      bf16* dst = sK + r * 136 + seg;
      stage16(src, dst);
      stage16(src + 16, dst + 16);
    }
    {  // stage V block transposed -> [d][key]
      int key = tid & 63, dseg = (tid >> 6) * 32;
      const bf16* src = v + (size_t)(kb0 + key) * D_ + dseg;
#pragma unroll
      for (int c = 0; c < 4; ++c) {
        v8bf val = *(const v8bf*)(src + c * 8);
#pragma unroll
        for (int j = 0; j < 8; ++j) sV[(dseg + c * 8 + j) * 72 + key] = val[j];
      }
    }
    async_drain();
    __syncthreads();

    // S = Q K^T  (16 q rows x 64 keys per wave)
    v8f sc[4];
#pragma unroll
    for (int j = 0; j < 4; ++j) sc[j] = vzero8();
#pragma unroll
    for (int j = 0; j < 4; ++j) {
      int n = j * 16 + ln;
#pragma unroll
      for (int c = 0; c < 4; ++c) {
        int kb = c * 32 + half * 8;
        v16bf bq = cat16(*(const v8bf*)(sK + n * 136 + kb),
                         *(const v8bf*)(sK + n * 136 + kb + 16));
        sc[j] = wmma_bf16(qf[c], bq, sc[j]);
      }
    }

    // scale + row max (rows r and r+8 split by lane-half)
    float rm[8];
#pragma unroll
    for (int r = 0; r < 8; ++r) {
      float mv = -3e30f;
#pragma unroll
      for (int j = 0; j < 4; ++j) { sc[j][r] *= scale; mv = fmaxf(mv, sc[j][r]); }
      for (int m = 1; m < 16; m <<= 1) mv = fmaxf(mv, __shfl_xor(mv, m, 32));
      rm[r] = mv;
    }
    if (ln == 0) {
#pragma unroll
      for (int r = 0; r < 8; ++r) {
        int row = r + 8 * half;
        float mo = sM[wid][row];
        float mn = fmaxf(mo, rm[r]);
        sMn[wid][row] = mn;
        sAl[wid][row] = __expf(mo - mn);
        sM[wid][row]  = mn;
      }
    }
    __syncthreads();

    // P = exp(S - m_new), row sums, P -> LDS in A-fragment-friendly layout
    float rs[8];
#pragma unroll
    for (int r = 0; r < 8; ++r) {
      float mn = sMn[wid][r + 8 * half];
      float sum = 0.f;
#pragma unroll
      for (int j = 0; j < 4; ++j) {
        float p = __expf(sc[j][r] - mn);
        sum += p;
        sP[wid][(r + 8 * half) * 72 + j * 16 + ln] = (bf16)p;
      }
      for (int m = 1; m < 16; m <<= 1) sum += __shfl_xor(sum, m, 32);
      rs[r] = sum;
    }
    if (ln == 0) {
#pragma unroll
      for (int r = 0; r < 8; ++r) {
        int row = r + 8 * half;
        sL[wid][row] = sL[wid][row] * sAl[wid][row] + rs[r];
      }
    }
    // rescale running O
#pragma unroll
    for (int r = 0; r < 8; ++r) {
      float al = sAl[wid][r + 8 * half];
#pragma unroll
      for (int dj = 0; dj < 8; ++dj) oacc[dj][r] *= al;
    }
    __syncthreads();

    // O += P @ V   (contraction over 64 keys = 2 WMMA K-steps)
    v16bf pf[2];
#pragma unroll
    for (int kc = 0; kc < 2; ++kc) {
      int kb = kc * 32 + half * 8;
      pf[kc] = cat16(*(const v8bf*)(&sP[wid][ln * 72 + kb]),
                     *(const v8bf*)(&sP[wid][ln * 72 + kb + 16]));
    }
#pragma unroll
    for (int dj = 0; dj < 8; ++dj) {
      v8f o = oacc[dj];
      int n = dj * 16 + ln;
#pragma unroll
      for (int kc = 0; kc < 2; ++kc) {
        int kb = kc * 32 + half * 8;
        v16bf vq = cat16(*(const v8bf*)(sV + n * 72 + kb),
                         *(const v8bf*)(sV + n * 72 + kb + 16));
        o = wmma_bf16(pf[kc], vq, o);
      }
      oacc[dj] = o;
    }
  }
  __syncthreads();

  // normalize + write to [token][INNER] bf16
#pragma unroll
  for (int r = 0; r < 8; ++r) {
    int row = r + 8 * half;
    float inv = 1.f / sL[wid][row];
    int qrow = qb * 128 + wid * 16 + row;
    size_t ob = ((size_t)(b * S_ + qrow)) * INNER_ + hh * D_;
#pragma unroll
    for (int dj = 0; dj < 8; ++dj)
      O[ob + dj * 16 + ln] = (bf16)(oacc[dj][r] * inv);
  }
}

// ---------------- fused up/gate GEMM + silu(u*silu(g)) -> bf16 ----------------
// BM=128, BN=64 per weight, BK=32; shares the A tile between both GEMMs.
// Double-buffered; A tile via async global->LDS.
__global__ __launch_bounds__(256) void gemm_upgate_kernel(const bf16* __restrict__ A,
                                                          const bf16* __restrict__ Wu,
                                                          const bf16* __restrict__ Wg,
                                                          bf16* __restrict__ gated,
                                                          int M, int N, int K) {
  const int tilesN = N >> 6;
  int tn = blockIdx.x % tilesN, tm = blockIdx.x / tilesN;
  int row0 = tm << 7, col0 = tn << 6;

  __shared__ __align__(16) bf16 sA[2][128 * 40];
  __shared__ __align__(16) bf16 sU[2][64 * 40];  // transposed [n][k]
  __shared__ __align__(16) bf16 sG[2][64 * 40];

  int tid = threadIdx.x;
  int lane = tid & 31, wid = tid >> 5;
  int wm = (wid >> 2) * 64, wn = (wid & 3) * 16;
  int ln = lane & 15, half = lane >> 4;

  v8f au[4], ag[4];
#pragma unroll
  for (int i = 0; i < 4; ++i) { au[i] = vzero8(); ag[i] = vzero8(); }

  auto stageA = [&](int k0, int buf) {
    int r = tid >> 1, seg = (tid & 1) * 16;
    stage16(A + (size_t)(row0 + r) * K + k0 + seg, &sA[buf][r * 40 + seg]);
  };
  auto stageW = [&](int k0, int buf) {
    int t = tid & 127;
    int kk = t & 31, nseg = ((t >> 5) & 3) * 16;
    const bf16* src = ((tid < 128) ? Wu : Wg) + (size_t)(k0 + kk) * N + col0 + nseg;
    bf16* dst = (tid < 128) ? sU[buf] : sG[buf];
    v8bf v0 = *(const v8bf*)(src);
    v8bf v1 = *(const v8bf*)(src + 8);
#pragma unroll
    for (int j = 0; j < 8; ++j) dst[(nseg + j) * 40 + kk]     = v0[j];
#pragma unroll
    for (int j = 0; j < 8; ++j) dst[(nseg + 8 + j) * 40 + kk] = v1[j];
  };

  const int nk = K >> 5;
  stageA(0, 0);
  stageW(0, 0);

  for (int kt = 0; kt < nk; ++kt) {
    int buf = kt & 1;
    async_drain();
    __syncthreads();
    if (kt + 1 < nk) {
      stageA((kt + 1) << 5, buf ^ 1);
      stageW((kt + 1) << 5, buf ^ 1);
    }

    v16bf af[4];
#pragma unroll
    for (int i = 0; i < 4; ++i) {
      int r = wm + i * 16 + ln;
      int kb = half * 8;
      af[i] = cat16(*(const v8bf*)(&sA[buf][r * 40 + kb]),
                    *(const v8bf*)(&sA[buf][r * 40 + 16 + kb]));
    }
    int n = wn + ln;
    int kb = half * 8;
    v16bf bu = cat16(*(const v8bf*)(&sU[buf][n * 40 + kb]),
                     *(const v8bf*)(&sU[buf][n * 40 + kb + 16]));
    v16bf bg = cat16(*(const v8bf*)(&sG[buf][n * 40 + kb]),
                     *(const v8bf*)(&sG[buf][n * 40 + kb + 16]));
#pragma unroll
    for (int i = 0; i < 4; ++i) {
      au[i] = wmma_bf16(af[i], bu, au[i]);
      ag[i] = wmma_bf16(af[i], bg, ag[i]);
    }
  }

#pragma unroll
  for (int i = 0; i < 4; ++i)
#pragma unroll
    for (int r = 0; r < 8; ++r) {
      int row = row0 + wm + i * 16 + r + 8 * half;
      int col = col0 + wn + ln;
      float u = au[i][r], g = ag[i][r];
      float sg = g / (1.f + __expf(-g));
      float z = u * sg;
      float out = z / (1.f + __expf(-z));
      gated[(size_t)row * N + col] = (bf16)out;
    }
}

// ---------------- launcher ----------------
extern "C" void kernel_launch(void* const* d_in, const int* in_sizes, int n_in,
                              void* d_out, int out_size, void* d_ws, size_t ws_size,
                              hipStream_t stream) {
  (void)in_sizes; (void)n_in; (void)out_size; (void)ws_size;
  const float* x      = (const float*)d_in[0];
  const float* cond   = (const float*)d_in[1];
  const float* pos    = (const float*)d_in[2];
  const float* w_ca   = (const float*)d_in[3];
  const float* w_qkv  = (const float*)d_in[4];
  const float* w_out  = (const float*)d_in[5];
  const float* w_cg   = (const float*)d_in[6];
  const float* w_up   = (const float*)d_in[7];
  const float* w_gate = (const float*)d_in[8];
  const float* w_down = (const float*)d_in[9];
  float* out = (float*)d_out;

  char* ws = (char*)d_ws;
  size_t off = 0;
  auto carve = [&](size_t bytes) -> char* {
    char* p = ws + off;
    off = (off + bytes + 255) & ~(size_t)255;
    return p;
  };

  float* condA  = (float*)carve((size_t)B_ * 2 * E_ * 4);
  float* condG  = (float*)carve((size_t)B_ * 2 * E_ * 4);
  bf16* h       = (bf16*)carve((size_t)NTOK * E_ * 2);
  bf16* qkv_raw = (bf16*)carve((size_t)NTOK * 3 * INNER_ * 2);  // reused as `gated`
  bf16* qa      = (bf16*)carve((size_t)NTOK * INNER_ * 2);
  bf16* ka      = (bf16*)carve((size_t)NTOK * INNER_ * 2);
  bf16* va      = (bf16*)carve((size_t)NTOK * INNER_ * 2);
  bf16* attn_o  = (bf16*)carve((size_t)NTOK * INNER_ * 2);
  float* x2     = (float*)carve((size_t)NTOK * E_ * 4);
  bf16* wqkv_b  = (bf16*)carve((size_t)E_ * 3 * INNER_ * 2);
  bf16* wout_b  = (bf16*)carve((size_t)INNER_ * E_ * 2);
  bf16* wup_b   = (bf16*)carve((size_t)E_ * FF_ * 2);
  bf16* wgate_b = (bf16*)carve((size_t)E_ * FF_ * 2);
  bf16* wdown_b = (bf16*)carve((size_t)FF_ * E_ * 2);
  bf16* gated   = qkv_raw;  // alias: qkv_raw dead after RoPE relayout

  // 1) weight precision conversion (fp32 -> bf16): compute-bound => bf16 WMMA path
  f32_to_bf16_kernel<<<2048, 256, 0, stream>>>(w_qkv,  wqkv_b,  E_ * 3 * INNER_);
  f32_to_bf16_kernel<<<1024, 256, 0, stream>>>(w_out,  wout_b,  INNER_ * E_);
  f32_to_bf16_kernel<<<1024, 256, 0, stream>>>(w_up,   wup_b,   E_ * FF_);
  f32_to_bf16_kernel<<<1024, 256, 0, stream>>>(w_gate, wgate_b, E_ * FF_);
  f32_to_bf16_kernel<<<1024, 256, 0, stream>>>(w_down, wdown_b, FF_ * E_);

  // 2) cond projections
  cond_gemm_kernel<<<128, 256, 0, stream>>>(cond, w_ca, w_cg, condA, condG);

  // 3) h1 = modulate(rms_norm(x), condA)
  rms_mod_kernel<<<NTOK, 256, 0, stream>>>(x, condA, h);

  // 4) qkv = h1 @ w_qkv     (8192x1024 @ 1024x6144, bf16 WMMA)
  gemm_bf16_kernel<0><<<(NTOK / 128) * (3 * INNER_ / 128), 256, 0, stream>>>(
      h, wqkv_b, qkv_raw, nullptr, NTOK, 3 * INNER_, E_);

  // 5) RoPE + relayout -> [B,H,S,D]
  rope_kernel<<<NTOK, 256, 0, stream>>>(qkv_raw, pos, qa, ka, va);

  // 6) flash attention (bf16 WMMA QK^T and PV, fp32 online softmax)
  attn_kernel<<<B_ * H_ * (S_ / 128), 256, 0, stream>>>(qa, ka, va, attn_o);

  // 7) x2 = attn_o @ w_out + x
  gemm_bf16_kernel<1><<<(NTOK / 128) * (E_ / 128), 256, 0, stream>>>(
      attn_o, wout_b, x2, x, NTOK, E_, INNER_);

  // 8) h2 = modulate(rms_norm(x2), condG)
  rms_mod_kernel<<<NTOK, 256, 0, stream>>>(x2, condG, h);

  // 9) gated = silu((h2@w_up) * silu(h2@w_gate))   (fused, shared A tile)
  gemm_upgate_kernel<<<(NTOK / 128) * (FF_ / 64), 256, 0, stream>>>(
      h, wup_b, wgate_b, gated, NTOK, FF_, E_);

  // 10) out = gated @ w_down + x2
  gemm_bf16_kernel<1><<<(NTOK / 128) * (E_ / 128), 256, 0, stream>>>(
      gated, wdown_b, out, x2, NTOK, E_, FF_);
}
